// MoE_47244640256434
// MI455X (gfx1250) — matrix operand order repaired and checked by reading, not compiled, not run on hardware
//
#include <hip/hip_runtime.h>
#include <hip/hip_bf16.h>
#include <math.h>

#define T_TOK   4096
#define D_MODEL 512
#define D_FF    2048
#define N_EXP   16
#define M_TILE  32            // tokens per block = two 16-row WMMA sub-tiles

typedef __attribute__((ext_vector_type(16))) __bf16 v16bf;
typedef __attribute__((ext_vector_type(8)))  float  v8f;

// ---------------- utility kernels ----------------

__global__ void moe_zero_kernel(float* p, int n) {
    int i = blockIdx.x * blockDim.x + threadIdx.x;
    if (i < n) p[i] = 0.0f;
}

// Grid-stride f32 -> bf16 conversion (one-time weight downconvert per launch).
__global__ void moe_cvt_bf16_kernel(const float* __restrict__ src,
                                    __bf16* __restrict__ dst, int n) {
    for (int i = blockIdx.x * blockDim.x + threadIdx.x; i < n;
         i += gridDim.x * blockDim.x)
        dst[i] = (__bf16)src[i];
}

// One wave32 per token: lanes 0..15 compute the 16 expert logits,
// lane 0 performs softmax + top-2 and writes the dense gate row.
__global__ void moe_router_kernel(const float* __restrict__ x,
                                  const float* __restrict__ rw,
                                  const float* __restrict__ rb,
                                  float* __restrict__ gates) {
    int tok  = (blockIdx.x * blockDim.x + threadIdx.x) >> 5;   // wave-uniform
    int lane = threadIdx.x & 31;
    if (tok >= T_TOK) return;

    const float* xr = x + (size_t)tok * D_MODEL;
    float logit = 0.0f;
    if (lane < N_EXP) {
        const float* wr = rw + (size_t)lane * D_MODEL;
        float acc = rb[lane];
        #pragma unroll 4
        for (int d = 0; d < D_MODEL; d += 4) {
            acc += xr[d] * wr[d] + xr[d + 1] * wr[d + 1]
                 + xr[d + 2] * wr[d + 2] + xr[d + 3] * wr[d + 3];
        }
        logit = acc;
    }

    float l[N_EXP];
    #pragma unroll
    for (int i = 0; i < N_EXP; ++i) l[i] = __shfl(logit, i, 32);

    if (lane == 0) {
        float m = l[0];
        #pragma unroll
        for (int i = 1; i < N_EXP; ++i) m = fmaxf(m, l[i]);
        float p[N_EXP], s = 0.0f;
        #pragma unroll
        for (int i = 0; i < N_EXP; ++i) { p[i] = expf(l[i] - m); s += p[i]; }
        float inv = 1.0f / s;
        #pragma unroll
        for (int i = 0; i < N_EXP; ++i) p[i] *= inv;

        int e1 = 0; float p1 = p[0];
        #pragma unroll
        for (int i = 1; i < N_EXP; ++i) if (p[i] > p1) { p1 = p[i]; e1 = i; }
        int e2 = -1; float p2 = -1.0f;
        #pragma unroll
        for (int i = 0; i < N_EXP; ++i)
            if (i != e1 && p[i] > p2) { p2 = p[i]; e2 = i; }

        float* gr = gates + (size_t)tok * N_EXP;
        #pragma unroll
        for (int i = 0; i < N_EXP; ++i)
            gr[i] = (i == e1) ? p1 : ((i == e2) ? p2 : 0.0f);
    }
}

// Deterministic, ordered compaction: one block per expert, serial scan.
__global__ void moe_compact_kernel(const float* __restrict__ gates,
                                   int* __restrict__ counts,
                                   int* __restrict__ tokens,
                                   float* __restrict__ gvals) {
    int e = blockIdx.x;
    if (threadIdx.x != 0) return;
    int c = 0;
    for (int t = 0; t < T_TOK; ++t) {
        float g = gates[(size_t)t * N_EXP + e];
        if (g > 0.0f) {
            tokens[(size_t)e * T_TOK + c] = t;
            gvals[(size_t)e * T_TOK + c]  = g;
            ++c;
        }
    }
    counts[e] = c;
}

// ---------------- WMMA fragment helpers (wave32, contiguous k = 16*(L>>4) + i in A and B) -------

// A (16x32 bf16) from LDS: lane m = L&15, K-half = L>>4.
__device__ __forceinline__ v16bf load_a_lds(const __bf16* __restrict__ X,
                                            int ldx, int k0, int lane) {
    int m  = lane & 15;
    int h16 = (lane >> 4) * 16;
    const __bf16* r = X + m * ldx + k0;
    v16bf a;
    #pragma unroll
    for (int i = 0; i < 16; ++i) {
        a[i] = r[h16 + i];
    }
    return a;
}

// B (32x16 bf16) from a row-major f32 weight [rows][ld]: B[k][n] = w[nb+n][k0+k].
__device__ __forceinline__ v16bf load_b_f32(const float* __restrict__ w,
                                            int ld, int nb, int k0, int lane) {
    int n    = lane & 15;
    int koff = k0 + (lane >> 4) * 16;
    const float* s = w + (size_t)(nb + n) * ld + koff;
    __builtin_prefetch(s + 32, 0, 1);   // global_prefetch_b8 on gfx1250
    v16bf b;
    #pragma unroll
    for (int i = 0; i < 16; ++i) b[i] = (__bf16)s[i];
    return b;
}

// Same fragment from a pre-converted bf16 weight: 32B-aligned direct load
// (two global_load_b128, no per-step conversion).
__device__ __forceinline__ v16bf load_b_bf16(const __bf16* __restrict__ w,
                                             int ld, int nb, int k0, int lane) {
    int n    = lane & 15;
    int koff = k0 + (lane >> 4) * 16;
    const __bf16* s = w + (size_t)(nb + n) * ld + koff;
    __builtin_prefetch(s + 64, 0, 1);
    return *(const v16bf*)s;
}

__device__ __forceinline__ float gelu_exact(float v) {
    return 0.5f * v * (1.0f + erff(v * 0.70710678118654752f));
}

// ---------------- fused expert FFN ----------------
// grid = (N_EXP, maxTiles), block = 256 (8 waves). Each block: 32 gathered
// tokens (two 16-row WMMA sub-tiles sharing every B fragment), full D_FF
// intermediate in LDS, WMMA bf16 for both GEMMs. WBF selects pre-converted
// bf16 weights (workspace) vs on-the-fly f32->bf16.

#define XLD (D_MODEL + 16)   // bf16 elements, padded for LDS bank spread
#define HLD (D_FF + 16)
#define XBYTES (M_TILE * XLD * 2)
#define HBYTES (M_TILE * HLD * 2)
#define SMEM_BYTES (XBYTES + HBYTES + M_TILE * 4 + M_TILE * 4)

template <bool WBF>
__global__ void moe_expert_kernel(const float*  __restrict__ x,
                                  const float*  __restrict__ w1f,
                                  const __bf16* __restrict__ w1h,
                                  const float*  __restrict__ b1,
                                  const float*  __restrict__ w2f,
                                  const __bf16* __restrict__ w2h,
                                  const float*  __restrict__ b2,
                                  const int*    __restrict__ counts,
                                  const int*    __restrict__ tokens,
                                  const float*  __restrict__ gvals,
                                  float* __restrict__ out) {
    extern __shared__ char smem[];
    __bf16* Xs     = (__bf16*)smem;
    __bf16* Hs     = (__bf16*)(smem + XBYTES);
    int*    tokids = (int*)  (smem + XBYTES + HBYTES);
    float*  gsh    = (float*)(tokids + M_TILE);

    const int e    = blockIdx.x;
    const int cnt  = counts[e];
    const int n0   = blockIdx.y * M_TILE;      // first token row of this tile
    if (n0 >= cnt) return;                     // block-uniform
    const int rows = min(M_TILE, cnt - n0);

    const int tid    = threadIdx.x;
    const int lane   = tid & 31;
    const int waveId = tid >> 5;

    const size_t woff = (size_t)e * D_FF * D_MODEL;
    const float*  w1fe = w1f + woff;
    const __bf16* w1he = w1h + woff;
    const float*  w2fe = w2f + woff;
    const __bf16* w2he = w2h + woff;
    const float*  b1e  = b1 + (size_t)e * D_FF;
    const float*  b2e  = b2 + (size_t)e * D_MODEL;

    if (tid < M_TILE) {
        int r = tid;
        tokids[r] = (r < rows) ? tokens[(size_t)e * T_TOK + n0 + r] : 0;
        gsh[r]    = (r < rows) ? gvals[(size_t)e * T_TOK + n0 + r]  : 0.0f;
    }
    __syncthreads();

    // gather X tile (32 x 512) -> bf16 LDS
    for (int idx = tid; idx < M_TILE * D_MODEL; idx += 256) {
        int r = idx >> 9;          // /512
        int d = idx & (D_MODEL - 1);
        float v = (r < rows) ? x[(size_t)tokids[r] * D_MODEL + d] : 0.0f;
        Xs[r * XLD + d] = (__bf16)v;
    }
    __syncthreads();

    // ---- GEMM1: H[32 x 2048] = gelu(X @ w1e^T + b1e), bf16 into LDS ----
    for (int tile = waveId; tile < D_FF / 16; tile += 8) {
        const int fbase = tile * 16;
        v8f c0 = {}, c1 = {};
        for (int k0 = 0; k0 < D_MODEL; k0 += 32) {
            v16bf b = WBF ? load_b_bf16(w1he, D_MODEL, fbase, k0, lane)
                          : load_b_f32 (w1fe, D_MODEL, fbase, k0, lane);
            v16bf a0 = load_a_lds(Xs,            XLD, k0, lane);
            v16bf a1 = load_a_lds(Xs + 16 * XLD, XLD, k0, lane);
            c0 = __builtin_amdgcn_wmma_f32_16x16x32_bf16(
                     false, a0, false, b, (short)0, c0, false, false);
            c1 = __builtin_amdgcn_wmma_f32_16x16x32_bf16(
                     false, a1, false, b, (short)0, c1, false, false);
        }
        const int   n    = lane & 15;
        const float bias = b1e[fbase + n];
        const int   mh   = (lane >> 4) * 8;
        #pragma unroll
        for (int r = 0; r < 8; ++r) {
            Hs[(mh + r) * HLD + fbase + n]      = (__bf16)gelu_exact(c0[r] + bias);
            Hs[(16 + mh + r) * HLD + fbase + n] = (__bf16)gelu_exact(c1[r] + bias);
        }
    }
    __syncthreads();

    // ---- GEMM2: Y[32 x 512] = H @ w2e^T + b2e, gate-scaled scatter ----
    for (int tile = waveId; tile < D_MODEL / 16; tile += 8) {
        const int nbase = tile * 16;
        v8f c0 = {}, c1 = {};
        for (int k0 = 0; k0 < D_FF; k0 += 32) {
            v16bf b = WBF ? load_b_bf16(w2he, D_FF, nbase, k0, lane)
                          : load_b_f32 (w2fe, D_FF, nbase, k0, lane);
            v16bf a0 = load_a_lds(Hs,            HLD, k0, lane);
            v16bf a1 = load_a_lds(Hs + 16 * HLD, HLD, k0, lane);
            c0 = __builtin_amdgcn_wmma_f32_16x16x32_bf16(
                     false, a0, false, b, (short)0, c0, false, false);
            c1 = __builtin_amdgcn_wmma_f32_16x16x32_bf16(
                     false, a1, false, b, (short)0, c1, false, false);
        }
        const int   n   = nbase + (lane & 15);
        const float b2v = b2e[n];
        const int   mh  = (lane >> 4) * 8;
        #pragma unroll
        for (int r = 0; r < 8; ++r) {
            int m0 = mh + r;
            int m1 = 16 + mh + r;
            if (m0 < rows) {
                float val = gsh[m0] * (c0[r] + b2v);
                atomicAdd(&out[(size_t)tokids[m0] * D_MODEL + n], val);
            }
            if (m1 < rows) {
                float val = gsh[m1] * (c1[r] + b2v);
                atomicAdd(&out[(size_t)tokids[m1] * D_MODEL + n], val);
            }
        }
    }
}

// ---------------- launcher ----------------

extern "C" void kernel_launch(void* const* d_in, const int* in_sizes, int n_in,
                              void* d_out, int out_size, void* d_ws, size_t ws_size,
                              hipStream_t stream) {
    const float* x   = (const float*)d_in[0];
    const float* rw  = (const float*)d_in[1];
    const float* rb  = (const float*)d_in[2];
    const float* w1  = (const float*)d_in[3];
    const float* b1  = (const float*)d_in[4];
    const float* w2  = (const float*)d_in[5];
    const float* b2  = (const float*)d_in[6];
    float* out = (float*)d_out;

    // workspace partition: routing state first, then optional bf16 weights
    const size_t W_ELEMS = (size_t)N_EXP * D_FF * D_MODEL;     // 16.7M per tensor
    char* ws = (char*)d_ws;
    int*   counts = (int*)ws;                                   ws += 256;
    int*   tokens = (int*)ws;                                   ws += (size_t)N_EXP * T_TOK * sizeof(int);
    float* gvals  = (float*)ws;                                 ws += (size_t)N_EXP * T_TOK * sizeof(float);
    float* gates  = (float*)ws;                                 ws += (size_t)T_TOK * N_EXP * sizeof(float);
    __bf16* w1h   = (__bf16*)ws;
    __bf16* w2h   = w1h + W_ELEMS;
    const size_t need = (size_t)((char*)(w2h + W_ELEMS) - (char*)d_ws);
    const bool useBf16W = ws_size >= need;                      // fixed per harness

    const int nOut = T_TOK * D_MODEL;
    moe_zero_kernel<<<(nOut + 255) / 256, 256, 0, stream>>>(out, nOut);

    // router: one wave32 per token, 4 waves per block
    moe_router_kernel<<<T_TOK / 4, 128, 0, stream>>>(x, rw, rb, gates);

    // deterministic compaction: one block per expert
    moe_compact_kernel<<<N_EXP, 32, 0, stream>>>(gates, counts, tokens, gvals);

    dim3 grid(N_EXP, T_TOK / M_TILE);
    if (useBf16W) {
        // one-time weight downconvert, then the cvt-free WMMA path
        moe_cvt_bf16_kernel<<<8192, 256, 0, stream>>>(w1, w1h, (int)W_ELEMS);
        moe_cvt_bf16_kernel<<<8192, 256, 0, stream>>>(w2, w2h, (int)W_ELEMS);
        moe_expert_kernel<true><<<grid, 256, SMEM_BYTES, stream>>>(
            x, w1, w1h, b1, w2, w2h, b2, counts, tokens, gvals, out);
    } else {
        moe_expert_kernel<false><<<grid, 256, SMEM_BYTES, stream>>>(
            x, w1, w1h, b1, w2, w2h, b2, counts, tokens, gvals, out);
    }
}
